// ModifiedWanTransformerBlock_87840671138312
// MI455X (gfx1250) — matrix operand order repaired and checked by reading, not compiled, not run on hardware
//
#include <hip/hip_runtime.h>
#include <hip/hip_bf16.h>
#include <math.h>

#define HEADS 12
#define DH 128
#define DIM 1536
#define FFN_DIM 8960
#define FN 8
#define PP 390
#define S_TOK 3120
#define S_PAD 3136        // padded to multiple of 32 for branch-free K/V staging
#define T_TOK 512
#define TOPK 4

typedef __attribute__((ext_vector_type(16))) __bf16 bf16x16;
typedef __attribute__((ext_vector_type(4)))  __bf16 bf16x4;
typedef __attribute__((ext_vector_type(8)))  float  f32x8;

__device__ __forceinline__ f32x8 f32x8_zero() {
  f32x8 z;
#pragma unroll
  for (int i = 0; i < 8; ++i) z[i] = 0.f;
  return z;
}

__device__ __forceinline__ f32x8 wmma_bf16(bf16x16 a, bf16x16 b, f32x8 c) {
  return __builtin_amdgcn_wmma_f32_16x16x32_bf16(false, a, false, b, (short)0, c,
                                                 false, false);
}

__device__ __forceinline__ bf16x4 cvt4(float4 v) {
  bf16x4 t;
  t[0] = (__bf16)v.x; t[1] = (__bf16)v.y; t[2] = (__bf16)v.z; t[3] = (__bf16)v.w;
  return t;
}

// tanh-approx GELU, branch-free: 0.5*x*(1+tanh(u)) == x * sigmoid(2u);
// uses hardware v_rcp_f32 instead of IEEE division.
__device__ __forceinline__ float gelu_tanh(float x) {
  float u2 = 1.5957691216057308f * (x + 0.044715f * x * x * x);
  return x * __builtin_amdgcn_rcpf(1.f + __expf(-u2));
}

// ---------- fragment helpers (contiguous k order k = 16*(lane>>4) + s, wave32) ----------
// A 16x32 frag from a contiguous f32 row: lane&15 = row (caller picks row ptr),
// lane>>4 selects K-halves {0-15} vs {16-31}. Two 8-float runs.
__device__ __forceinline__ bf16x16 cvt_a_frag(const float* row, int k0) {
  int h0 = (threadIdx.x >> 4) & 1;
  const float4* p0 = (const float4*)(row + k0 + h0 * 16);
  const float4* p1 = (const float4*)(row + k0 + h0 * 16 + 8);
  float4 a0 = p0[0], a1 = p0[1], b0 = p1[0], b1 = p1[1];
  bf16x16 a;
  a[0] = (__bf16)a0.x;  a[1] = (__bf16)a0.y;  a[2] = (__bf16)a0.z;  a[3] = (__bf16)a0.w;
  a[4] = (__bf16)a1.x;  a[5] = (__bf16)a1.y;  a[6] = (__bf16)a1.z;  a[7] = (__bf16)a1.w;
  a[8] = (__bf16)b0.x;  a[9] = (__bf16)b0.y;  a[10] = (__bf16)b0.z; a[11] = (__bf16)b0.w;
  a[12] = (__bf16)b1.x; a[13] = (__bf16)b1.y; a[14] = (__bf16)b1.z; a[15] = (__bf16)b1.w;
  return a;
}
// 16 contiguous f32 -> bf16x16, scaled (scale=0 kills invalid lanes branch-free)
__device__ __forceinline__ bf16x16 cvt16_scaled(const float* p, float s) {
  const float4* q = (const float4*)p;
  float4 v0 = q[0], v1 = q[1], v2 = q[2], v3 = q[3];
  bf16x16 b;
  b[0] = (__bf16)(v0.x * s);  b[1] = (__bf16)(v0.y * s);
  b[2] = (__bf16)(v0.z * s);  b[3] = (__bf16)(v0.w * s);
  b[4] = (__bf16)(v1.x * s);  b[5] = (__bf16)(v1.y * s);
  b[6] = (__bf16)(v1.z * s);  b[7] = (__bf16)(v1.w * s);
  b[8] = (__bf16)(v2.x * s);  b[9] = (__bf16)(v2.y * s);
  b[10] = (__bf16)(v2.z * s); b[11] = (__bf16)(v2.w * s);
  b[12] = (__bf16)(v3.x * s); b[13] = (__bf16)(v3.y * s);
  b[14] = (__bf16)(v3.z * s); b[15] = (__bf16)(v3.w * s);
  return b;
}
// 16 contiguous bf16 from LDS -> frag (2x ds_load_b128)
__device__ __forceinline__ bf16x16 lds16(const __bf16* p) {
  bf16x16 b;
#pragma unroll
  for (int e = 0; e < 16; ++e) b[e] = p[e];
  return b;
}
// A frag from LDS tile (row-major, ld elems per row)
__device__ __forceinline__ bf16x16 lds_a_frag(const __bf16* sm, int ld, int row0) {
  int lane = threadIdx.x & 31;
  int r = lane & 15, h0 = lane >> 4;
  const __bf16* base = sm + (size_t)(row0 + r) * ld;
  bf16x16 a;
#pragma unroll
  for (int e = 0; e < 8; ++e) a[e] = base[h0 * 16 + e];
#pragma unroll
  for (int e = 0; e < 8; ++e) a[8 + e] = base[h0 * 16 + 8 + e];
  return a;
}

// =========================== generic GEMM ===========================
// C(MxN) = act(A(MxK) @ W(KxN) + bias). Block 128 thr = 4 waves, tile 64x128,
// BK=32; each wave owns 32x64 (2x4 WMMA tiles, 8 wmma/K-step).
// M-edge handled by CLAMPED loads (no exec-mask divergence); stores use a
// uniform full-tile fast path, guarded path only on the edge block row.
// ACT is a compile-time flag so epilogues are straight-line code.
#define BM 64
#define BN 128
#define BK 32
template <int ACT>
__global__ void __launch_bounds__(128)
k_gemm(const float* __restrict__ A, const float* __restrict__ W,
       const float* __restrict__ bias, float* __restrict__ C,
       int M, int N, int K) {
  __shared__ __bf16 As[BM * BK];     // [row][k]
  __shared__ __bf16 Bt[BN * BK];     // transposed: [col][k]
  const int m0 = blockIdx.y * BM, n0 = blockIdx.x * BN;
  const int tid = threadIdx.x, lane = tid & 31, wave = tid >> 5;
  const int wm = (wave >> 1) * 32, wn = (wave & 1) * 64;
  f32x8 acc[2][4];
#pragma unroll
  for (int i = 0; i < 2; ++i)
#pragma unroll
    for (int j = 0; j < 4; ++j) acc[i][j] = f32x8_zero();

  for (int k0 = 0; k0 < K; k0 += BK) {
#pragma unroll
    for (int i = 0; i < 4; ++i) {
      int idx = tid + i * 128;
      int r = idx >> 3, c4 = (idx & 7) * 4;
      int gr = m0 + r; gr = gr < M ? gr : M - 1;   // clamp, never branch
      float4 v = *(const float4*)(A + (size_t)gr * K + k0 + c4);
      *(bf16x4*)(&As[r * BK + c4]) = cvt4(v);
    }
#pragma unroll
    for (int i = 0; i < 8; ++i) {
      int idx = tid + i * 128;
      int r = idx >> 5, c4 = (idx & 31) * 4;   // r: k row 0..31, c4: col
      float4 v = *(const float4*)(W + (size_t)(k0 + r) * N + n0 + c4);
      Bt[(c4 + 0) * BK + r] = (__bf16)v.x;
      Bt[(c4 + 1) * BK + r] = (__bf16)v.y;
      Bt[(c4 + 2) * BK + r] = (__bf16)v.z;
      Bt[(c4 + 3) * BK + r] = (__bf16)v.w;
    }
    __syncthreads();
    const int cn = lane & 15, h0 = lane >> 4;
    bf16x16 af[2], bfv[4];
#pragma unroll
    for (int i = 0; i < 2; ++i) af[i] = lds_a_frag(As, BK, wm + i * 16);
#pragma unroll
    for (int j = 0; j < 4; ++j)
      bfv[j] = lds16(&Bt[(wn + j * 16 + cn) * BK + h0 * 16]);
#pragma unroll
    for (int i = 0; i < 2; ++i)
#pragma unroll
      for (int j = 0; j < 4; ++j) acc[i][j] = wmma_bf16(af[i], bfv[j], acc[i][j]);
    __syncthreads();
  }
  const int h0 = lane >> 4, cn = lane & 15;
  const bool full = (m0 + BM) <= M;          // uniform across block
  if (full) {
#pragma unroll
    for (int i = 0; i < 2; ++i)
#pragma unroll
      for (int j = 0; j < 4; ++j) {
        int colg = n0 + wn + j * 16 + cn;
        float bv = bias[colg];
#pragma unroll
        for (int p = 0; p < 8; ++p) {
          int rowg = m0 + wm + i * 16 + p + 8 * h0;
          float v = acc[i][j][p] + bv;
          if (ACT == 1) v = gelu_tanh(v);
          C[(size_t)rowg * N + colg] = v;
        }
      }
  } else {
#pragma unroll
    for (int i = 0; i < 2; ++i)
#pragma unroll
      for (int j = 0; j < 4; ++j) {
        int colg = n0 + wn + j * 16 + cn;
        float bv = bias[colg];
#pragma unroll
        for (int p = 0; p < 8; ++p) {
          int rowg = m0 + wm + i * 16 + p + 8 * h0;
          if (rowg < M) {
            float v = acc[i][j][p] + bv;
            if (ACT == 1) v = gelu_tanh(v);
            C[(size_t)rowg * N + colg] = v;
          }
        }
      }
  }
}

// =========================== flash attention ===========================
// Q (Sq,H,DH), K/V ((Skv padded to mult 32),H,DH) f32; O (Sq,H,DH).
// One wave per 16 query rows; K/V staged per 32-key chunk in LDS.
__global__ void __launch_bounds__(32)
k_flash(const float* __restrict__ Q, const float* __restrict__ Kv,
        const float* __restrict__ Vv, float* __restrict__ O,
        int Skv, float scale) {
  __shared__ __bf16 Kl[32 * DH];       // [kv][d]
  __shared__ __bf16 Vl[DH * 32];       // [d][kv] (transposed)
  __shared__ float Sld[16][32];
  __shared__ float rowm[16], rowl[16], rowf[16];
  const int h = blockIdx.y;
  const int q0 = blockIdx.x * 16;
  const int lane = threadIdx.x;
  const int r = lane & 15, h0 = lane >> 4, cn = lane & 15;
  const int ld = HEADS * DH;

  bf16x16 qf[4];
  {
    const float* qrow = Q + (size_t)(q0 + r) * ld + h * DH;
#pragma unroll
    for (int ks = 0; ks < 4; ++ks) qf[ks] = cvt_a_frag(qrow, ks * 32);
  }
  f32x8 oacc[8];
#pragma unroll
  for (int j = 0; j < 8; ++j) oacc[j] = f32x8_zero();
  if (lane < 16) { rowm[lane] = -1e30f; rowl[lane] = 0.f; }
  __syncthreads();

  for (int kv0 = 0; kv0 < Skv; kv0 += 32) {
    // ---- stage K/V chunk: lane owns one kv row (rows are zero-padded) ----
    {
      const float* krow = Kv + (size_t)(kv0 + lane) * ld + h * DH;
      const float* vrow = Vv + (size_t)(kv0 + lane) * ld + h * DH;
#pragma unroll 8
      for (int d = 0; d < DH; d += 4) {
        float4 kf = *(const float4*)(krow + d);
        float4 vf = *(const float4*)(vrow + d);
        *(bf16x4*)(&Kl[lane * DH + d]) = cvt4(kf);
        Vl[(d + 0) * 32 + lane] = (__bf16)vf.x;
        Vl[(d + 1) * 32 + lane] = (__bf16)vf.y;
        Vl[(d + 2) * 32 + lane] = (__bf16)vf.z;
        Vl[(d + 3) * 32 + lane] = (__bf16)vf.w;
      }
    }
    __syncthreads();
    // ---- S tile = Q @ K^T (two 16-col halves) ----
#pragma unroll
    for (int nh = 0; nh < 2; ++nh) {
      f32x8 sacc = f32x8_zero();
      const __bf16* kbase = &Kl[(nh * 16 + cn) * DH];
#pragma unroll
      for (int ks = 0; ks < 4; ++ks)
        sacc = wmma_bf16(qf[ks], lds16(kbase + ks * 32 + h0 * 16), sacc);
      bool valid = (kv0 + nh * 16 + cn) < Skv;
#pragma unroll
      for (int p = 0; p < 8; ++p)
        Sld[p + 8 * h0][nh * 16 + cn] = valid ? sacc[p] * scale : -1e30f;
    }
    __syncthreads();
    // ---- online softmax (2 lanes per row) ----
    {
      int rr = lane >> 1, hh = lane & 1;
      float mx = -1e30f;
#pragma unroll
      for (int c = 0; c < 16; ++c) mx = fmaxf(mx, Sld[rr][hh * 16 + c]);
      mx = fmaxf(mx, __shfl_xor(mx, 1, 32));
      float mprev = rowm[rr];
      float mnew = fmaxf(mprev, mx);
      float sum = 0.f;
#pragma unroll
      for (int c = 0; c < 16; ++c) {
        float p = __expf(Sld[rr][hh * 16 + c] - mnew);
        Sld[rr][hh * 16 + c] = p;
        sum += p;
      }
      sum += __shfl_xor(sum, 1, 32);
      if (hh == 0) {
        rowf[rr] = __expf(mprev - mnew);
        rowl[rr] = rowl[rr] * rowf[rr] + sum;
        rowm[rr] = mnew;
      }
    }
    __syncthreads();
#pragma unroll
    for (int j = 0; j < 8; ++j)
#pragma unroll
      for (int p = 0; p < 8; ++p) oacc[j][p] *= rowf[p + 8 * h0];
    // ---- P fragment (16x32 from LDS, contiguous f32 runs) ----
    bf16x16 pf;
#pragma unroll
    for (int e = 0; e < 8; ++e) pf[e] = (__bf16)Sld[r][h0 * 16 + e];
#pragma unroll
    for (int e = 0; e < 8; ++e) pf[8 + e] = (__bf16)Sld[r][h0 * 16 + 8 + e];
    // ---- O += P @ V (V frags contiguous in transposed LDS) ----
#pragma unroll
    for (int j = 0; j < 8; ++j)
      oacc[j] = wmma_bf16(pf, lds16(&Vl[(j * 16 + cn) * 32 + h0 * 16]), oacc[j]);
    __syncthreads();
  }
  float invl[8];
#pragma unroll
  for (int p = 0; p < 8; ++p) invl[p] = 1.f / rowl[p + 8 * h0];
#pragma unroll
  for (int j = 0; j < 8; ++j)
#pragma unroll
    for (int p = 0; p < 8; ++p)
      O[(size_t)(q0 + p + 8 * h0) * ld + h * DH + j * 16 + cn] = oacc[j][p] * invl[p];
}

// ============ qk logits: Lg[h,t,p] = dot(K2[t,h,:], Q[p,h,:]) * scale ============
__global__ void __launch_bounds__(32)
k_qk_logits(const float* __restrict__ K2, const float* __restrict__ Qm,
            float* __restrict__ Lg, float scale) {
  const int h = blockIdx.z;
  const int t0 = blockIdx.y * 16, p0 = blockIdx.x * 16;
  const int lane = threadIdx.x, r = lane & 15, h0 = lane >> 4, cn = lane & 15;
  const int ld = HEADS * DH;
  const float* arow = K2 + (size_t)(t0 + r) * ld + h * DH;
  const float* brow = Qm + (size_t)(p0 + cn) * ld + h * DH;
  f32x8 acc = f32x8_zero();
#pragma unroll
  for (int ks = 0; ks < 4; ++ks)
    acc = wmma_bf16(cvt_a_frag(arow, ks * 32),
                    cvt16_scaled(brow + ks * 32 + h0 * 16, 1.f), acc);
#pragma unroll
  for (int p = 0; p < 8; ++p)
    Lg[((size_t)h * T_TOK + t0 + p + 8 * h0) * S_TOK + p0 + cn] = acc[p] * scale;
}

// ==== imap: IM[h,t,f*PP+p] = dot(A4[f*PP+vid(h,t,f),h,:], A4[f*PP+p,h,:]) ====
__global__ void __launch_bounds__(32)
k_imap(const float* __restrict__ A4, const int* __restrict__ vidx,
       float* __restrict__ IM) {
  const int hf = blockIdx.z;
  const int h = hf >> 3, f = hf & 7;
  const int t0 = blockIdx.y * 16, p0 = blockIdx.x * 16;
  const int lane = threadIdx.x, r = lane & 15, h0 = lane >> 4, cn = lane & 15;
  const int ld = HEADS * DH;
  const int t = t0 + r;
  const int vs = vidx[((size_t)h * T_TOK + t) * FN + f];
  const float* arow = A4 + (size_t)(f * PP + vs) * ld + h * DH;
  const int pc = p0 + cn;
  const bool pv = pc < PP;
  const float pm = pv ? 1.f : 0.f;
  const float* brow = A4 + (size_t)(f * PP + (pv ? pc : 0)) * ld + h * DH;
  f32x8 acc = f32x8_zero();
#pragma unroll
  for (int ks = 0; ks < 4; ++ks)
    acc = wmma_bf16(cvt_a_frag(arow, ks * 32),
                    cvt16_scaled(brow + ks * 32 + h0 * 16, pm), acc);
  if (pv) {
#pragma unroll
    for (int p = 0; p < 8; ++p)
      IM[((size_t)h * T_TOK + t0 + p + 8 * h0) * S_TOK + f * PP + pc] = acc[p];
  }
}

// =========================== elementwise / reductions ===========================
__global__ void k_mods(const float* __restrict__ sst, const float* __restrict__ temb,
                       float* __restrict__ mods) {
  int i = blockIdx.x * 256 + threadIdx.x;
  if (i < 6 * DIM) mods[i] = sst[i] + temb[i];
}

__global__ void k_zero4(float4* __restrict__ p, int n4) {
  int i = blockIdx.x * 256 + threadIdx.x;
  if (i < n4) p[i] = make_float4(0.f, 0.f, 0.f, 0.f);
}

__global__ void __launch_bounds__(256)
k_ln(const float* __restrict__ x, float* __restrict__ y,
     const float* __restrict__ w, const float* __restrict__ b,
     const float* __restrict__ srow, const float* __restrict__ shrow) {
  __shared__ float red[256];
  const int row = blockIdx.x, tid = threadIdx.x;
  const float* xr = x + (size_t)row * DIM;
  float vbuf[6];
  float s = 0.f;
#pragma unroll
  for (int i = 0; i < 6; ++i) { vbuf[i] = xr[tid + i * 256]; s += vbuf[i]; }
  red[tid] = s; __syncthreads();
  for (int st = 128; st > 0; st >>= 1) { if (tid < st) red[tid] += red[tid + st]; __syncthreads(); }
  float mean = red[0] * (1.f / DIM); __syncthreads();
  s = 0.f;
#pragma unroll
  for (int i = 0; i < 6; ++i) { float d = vbuf[i] - mean; s += d * d; }
  red[tid] = s; __syncthreads();
  for (int st = 128; st > 0; st >>= 1) { if (tid < st) red[tid] += red[tid + st]; __syncthreads(); }
  float inv = rsqrtf(red[0] * (1.f / DIM) + 1e-6f);
  float* yr = y + (size_t)row * DIM;
#pragma unroll
  for (int i = 0; i < 6; ++i) {
    int c = tid + i * 256;
    float v = (vbuf[i] - mean) * inv;
    if (w) v = v * w[c] + b[c];
    if (srow) v = v * (1.f + srow[c]) + shrow[c];
    yr[c] = v;
  }
}

__global__ void __launch_bounds__(256)
k_rms(float* __restrict__ x, const float* __restrict__ g) {
  __shared__ float red[256];
  const int row = blockIdx.x, tid = threadIdx.x;
  float* xr = x + (size_t)row * DIM;
  float vbuf[6];
  float s = 0.f;
#pragma unroll
  for (int i = 0; i < 6; ++i) { vbuf[i] = xr[tid + i * 256]; s += vbuf[i] * vbuf[i]; }
  red[tid] = s; __syncthreads();
  for (int st = 128; st > 0; st >>= 1) { if (tid < st) red[tid] += red[tid + st]; __syncthreads(); }
  float inv = rsqrtf(red[0] * (1.f / DIM) + 1e-6f);
#pragma unroll
  for (int i = 0; i < 6; ++i) {
    int c = tid + i * 256;
    xr[c] = vbuf[i] * inv * g[c];
  }
}

__global__ void k_rope(float* __restrict__ x, const float* __restrict__ fc,
                       const float* __restrict__ fs) {
  int idx = blockIdx.x * 256 + threadIdx.x;
  if (idx >= S_TOK * HEADS * (DH / 2)) return;
  int i = idx & 63;
  int h = (idx >> 6) % HEADS;
  int s = idx / (64 * HEADS);
  float* p = x + (size_t)s * DIM + h * DH + 2 * i;
  float c = fc[(size_t)s * DH + 2 * i];
  float sn = fs[(size_t)s * DH + 2 * i + 1];
  float x1 = p[0], x2 = p[1];
  p[0] = x1 * c - x2 * sn;
  p[1] = x1 * sn + x2 * c;
}

// out = base + add * gate[col] (gate nullable -> 1); float4 over S*DIM
__global__ void k_add_gate(const float4* __restrict__ base, const float4* __restrict__ add,
                           const float* __restrict__ gate, float4* __restrict__ out, int n4) {
  int i = blockIdx.x * 256 + threadIdx.x;
  if (i >= n4) return;
  float4 bv = base[i], av = add[i];
  float4 g = make_float4(1.f, 1.f, 1.f, 1.f);
  if (gate) {
    int c = (i * 4) % DIM;
    g = *(const float4*)(gate + c);
  }
  out[i] = make_float4(bv.x + av.x * g.x, bv.y + av.y * g.y,
                       bv.z + av.z * g.z, bv.w + av.w * g.w);
}

__global__ void k_argmax(const float* __restrict__ Lg, int* __restrict__ vidx) {
  int idx = blockIdx.x * 256 + threadIdx.x;
  if (idx >= HEADS * T_TOK * FN) return;
  int f = idx % FN;
  int t = (idx / FN) % T_TOK;
  int h = idx / (FN * T_TOK);
  const float* row = Lg + ((size_t)h * T_TOK + t) * S_TOK + f * PP;
  float best = row[0]; int bi = 0;
  for (int p = 1; p < PP; ++p) { float v = row[p]; if (v > best) { best = v; bi = p; } }
  vidx[idx] = bi;
}

__global__ void __launch_bounds__(256)
k_cross_softmax(const float* __restrict__ Lg, float* __restrict__ cm) {
  __shared__ float red[256];
  const int p = blockIdx.x, tid = threadIdx.x;
  float acc0 = 0.f, acc1 = 0.f;
  for (int h = 0; h < HEADS; ++h) {
    const float* col = Lg + (size_t)h * T_TOK * S_TOK + p;
    float v0 = col[(size_t)tid * S_TOK];
    float v1 = col[(size_t)(tid + 256) * S_TOK];
    red[tid] = fmaxf(v0, v1); __syncthreads();
    for (int st = 128; st > 0; st >>= 1) { if (tid < st) red[tid] = fmaxf(red[tid], red[tid + st]); __syncthreads(); }
    float m = red[0]; __syncthreads();
    float e0 = __expf(v0 - m), e1 = __expf(v1 - m);
    red[tid] = e0 + e1; __syncthreads();
    for (int st = 128; st > 0; st >>= 1) { if (tid < st) red[tid] += red[tid + st]; __syncthreads(); }
    float inv = 1.f / red[0]; __syncthreads();
    acc0 += e0 * inv; acc1 += e1 * inv;
  }
  cm[(size_t)tid * S_TOK + p] = acc0 * (1.f / HEADS);
  cm[(size_t)(tid + 256) * S_TOK + p] = acc1 * (1.f / HEADS);
}

__global__ void k_energy(const float* __restrict__ A4, float* __restrict__ E) {
  int idx = blockIdx.x * 256 + threadIdx.x;
  if (idx >= HEADS * S_TOK) return;
  int h = idx / S_TOK, s = idx % S_TOK;
  const float4* r = (const float4*)(A4 + (size_t)s * DIM + h * DH);
  float acc = 0.f;
#pragma unroll 8
  for (int d = 0; d < DH / 4; ++d) {
    float4 v = r[d];
    acc += v.x * v.x + v.y * v.y + v.z * v.z + v.w * v.w;
  }
  E[idx] = sqrtf(acc);
}

__global__ void __launch_bounds__(256)
k_headscore(const float* __restrict__ E, int* __restrict__ flags) {
  __shared__ float red[256], red2[256];
  __shared__ float scores[HEADS];
  const int tid = threadIdx.x;
  for (int h = 0; h < HEADS; ++h) {
    float s = 0.f, s2 = 0.f;
    for (int i = tid; i < S_TOK; i += 256) {
      float v = E[h * S_TOK + i];
      s += v; s2 += v * v;
    }
    red[tid] = s; red2[tid] = s2; __syncthreads();
    for (int st = 128; st > 0; st >>= 1) {
      if (tid < st) { red[tid] += red[tid + st]; red2[tid] += red2[tid + st]; }
      __syncthreads();
    }
    if (tid == 0) {
      float mean = red[0] / S_TOK;
      float var = fmaxf(red2[0] / S_TOK - mean * mean, 0.f);
      scores[h] = sqrtf(var) / (mean + 1e-6f);
    }
    __syncthreads();
  }
  if (tid == 0) {
    bool used[HEADS] = {};
    for (int k = 0; k < TOPK; ++k) {
      float best = -1e30f; int bi = 0;
      for (int h = 0; h < HEADS; ++h)
        if (!used[h] && scores[h] > best) { best = scores[h]; bi = h; }
      used[bi] = true;
    }
    for (int h = 0; h < HEADS; ++h) flags[h] = used[h] ? 1 : 0;
  }
}

__global__ void k_imap_norm(float* __restrict__ IM) {
  int idx = blockIdx.x * 256 + threadIdx.x;
  if (idx >= HEADS * S_TOK) return;
  int h = idx / S_TOK, s = idx % S_TOK;
  float* col = IM + (size_t)h * T_TOK * S_TOK + s;
  float sum = 0.f, sumsq = 0.f;
  for (int t = 0; t < T_TOK; ++t) { float v = col[(size_t)t * S_TOK]; sum += v; sumsq += v * v; }
  float mean = sum / T_TOK;
  float var = fmaxf((sumsq - T_TOK * mean * mean) / (T_TOK - 1), 0.f);
  float inv = 1.f / (sqrtf(var) + 1e-6f);
  for (int t = 0; t < T_TOK; ++t) {
    size_t o = (size_t)t * S_TOK;
    col[o] = (col[o] - mean) * inv;
  }
}

__global__ void k_imap_reduce(const float* __restrict__ IM, const int* __restrict__ flags,
                              float* __restrict__ out_sel, float* __restrict__ out_sum) {
  int idx = blockIdx.x * 256 + threadIdx.x;
  if (idx >= T_TOK * S_TOK) return;
  int t = idx / S_TOK, s = idx % S_TOK;
  float ssum = 0.f, ssel = 0.f;
#pragma unroll
  for (int h = 0; h < HEADS; ++h) {
    float v = IM[((size_t)h * T_TOK + t) * S_TOK + s];
    ssum += v;
    if (flags[h]) ssel += v;
  }
  out_sel[idx] = ssel * (1.f / TOPK);
  out_sum[idx] = ssum;
}

// =========================== host orchestration ===========================
static inline dim3 gemm_grid(int M, int N) { return dim3((N + BN - 1) / BN, (M + BM - 1) / BM); }

extern "C" void kernel_launch(void* const* d_in, const int* in_sizes, int n_in,
                              void* d_out, int out_size, void* d_ws, size_t ws_size,
                              hipStream_t stream) {
  const float* hidden = (const float*)d_in[0];
  const float* enc    = (const float*)d_in[1];
  const float* temb   = (const float*)d_in[2];
  const float* fcos   = (const float*)d_in[3];
  const float* fsin   = (const float*)d_in[4];
  const float* sst    = (const float*)d_in[5];
  const float* a1_wq = (const float*)d_in[6],  *a1_bq = (const float*)d_in[7];
  const float* a1_wk = (const float*)d_in[8],  *a1_bk = (const float*)d_in[9];
  const float* a1_wv = (const float*)d_in[10], *a1_bv = (const float*)d_in[11];
  const float* a1_wo = (const float*)d_in[12], *a1_bo = (const float*)d_in[13];
  const float* a1_gq = (const float*)d_in[14], *a1_gk = (const float*)d_in[15];
  const float* a2_wq = (const float*)d_in[16], *a2_bq = (const float*)d_in[17];
  const float* a2_wk = (const float*)d_in[18], *a2_bk = (const float*)d_in[19];
  const float* a2_wv = (const float*)d_in[20], *a2_bv = (const float*)d_in[21];
  const float* a2_wo = (const float*)d_in[22], *a2_bo = (const float*)d_in[23];
  const float* a2_gq = (const float*)d_in[24], *a2_gk = (const float*)d_in[25];
  const float* n2_w  = (const float*)d_in[26], *n2_b  = (const float*)d_in[27];
  const float* ffw1  = (const float*)d_in[28], *ffb1  = (const float*)d_in[29];
  const float* ffw2  = (const float*)d_in[30], *ffb2  = (const float*)d_in[31];

  const size_t SD  = (size_t)S_TOK * DIM;
  const size_t SPD = (size_t)S_PAD * DIM;
  const size_t TD  = (size_t)T_TOK * DIM;
  const size_t HTS = (size_t)HEADS * T_TOK * S_TOK;

  float* w = (float*)d_ws;
  size_t o = 0;
  float* mods = w + o;  o += 6 * DIM;
  float* xb   = w + o;  o += SD;
  float* qb   = w + o;  o += SD;
  float* kb   = w + o;  o += SPD;   // padded (zero tail) for branch-free staging
  float* vb   = w + o;  o += SPD;
  float* atb  = w + o;  o += SD;
  float* hsb  = w + o;  o += SD;
  float* tmpb = w + o;  o += SD;
  float* a4b  = w + o;  o += SD;
  float* k2b  = w + o;  o += TD;
  float* v2b  = w + o;  o += TD;
  float* Eb   = w + o;  o += (size_t)HEADS * S_TOK;
  int*   flags = (int*)(w + o); o += 16;
  int*   vidx  = (int*)(w + o); o += (size_t)HEADS * T_TOK * FN;
  float* logits = w + o;
  float* imap   = logits + HTS;
  float* ffnmid = logits;           // reused after analysis outputs are produced
  (void)ws_size; (void)n_in; (void)in_sizes; (void)out_size;

  float* out_hs  = (float*)d_out;
  float* out_sel = out_hs + SD;
  float* out_sum = out_sel + (size_t)T_TOK * S_TOK;
  float* out_cm  = out_sum + (size_t)T_TOK * S_TOK;

  const float scale = 0.08838834764831843f;  // 1/sqrt(128)
  const int nSD4 = (int)(SD / 4);
  const int nTail4 = (S_PAD - S_TOK) * DIM / 4;

  k_mods<<<(6 * DIM + 255) / 256, 256, 0, stream>>>(sst, temb, mods);
  k_ln<<<S_TOK, 256, 0, stream>>>(hidden, xb, nullptr, nullptr, mods + 1 * DIM, mods + 0 * DIM);
  // self-attn projections (+ zero pad rows of K/V)
  k_gemm<0><<<gemm_grid(S_TOK, DIM), 128, 0, stream>>>(xb, a1_wq, a1_bq, qb, S_TOK, DIM, DIM);
  k_gemm<0><<<gemm_grid(S_TOK, DIM), 128, 0, stream>>>(xb, a1_wk, a1_bk, kb, S_TOK, DIM, DIM);
  k_gemm<0><<<gemm_grid(S_TOK, DIM), 128, 0, stream>>>(xb, a1_wv, a1_bv, vb, S_TOK, DIM, DIM);
  k_zero4<<<(nTail4 + 255) / 256, 256, 0, stream>>>((float4*)(kb + SD), nTail4);
  k_zero4<<<(nTail4 + 255) / 256, 256, 0, stream>>>((float4*)(vb + SD), nTail4);
  k_rms<<<S_TOK, 256, 0, stream>>>(qb, a1_gq);
  k_rms<<<S_TOK, 256, 0, stream>>>(kb, a1_gk);
  k_rope<<<(S_TOK * HEADS * 64 + 255) / 256, 256, 0, stream>>>(qb, fcos, fsin);
  k_rope<<<(S_TOK * HEADS * 64 + 255) / 256, 256, 0, stream>>>(kb, fcos, fsin);
  k_flash<<<dim3(S_TOK / 16, HEADS), 32, 0, stream>>>(qb, kb, vb, atb, S_TOK, scale);
  k_gemm<0><<<gemm_grid(S_TOK, DIM), 128, 0, stream>>>(atb, a1_wo, a1_bo, tmpb, S_TOK, DIM, DIM);
  k_add_gate<<<(nSD4 + 255) / 256, 256, 0, stream>>>((const float4*)hidden, (const float4*)tmpb,
                                                     mods + 2 * DIM, (float4*)hsb, nSD4);
  // cross attention
  k_ln<<<S_TOK, 256, 0, stream>>>(hsb, xb, n2_w, n2_b, nullptr, nullptr);
  k_gemm<0><<<gemm_grid(S_TOK, DIM), 128, 0, stream>>>(xb, a2_wq, a2_bq, qb, S_TOK, DIM, DIM);
  k_rms<<<S_TOK, 256, 0, stream>>>(qb, a2_gq);
  k_gemm<0><<<gemm_grid(T_TOK, DIM), 128, 0, stream>>>(enc, a2_wk, a2_bk, k2b, T_TOK, DIM, DIM);
  k_rms<<<T_TOK, 256, 0, stream>>>(k2b, a2_gk);
  k_gemm<0><<<gemm_grid(T_TOK, DIM), 128, 0, stream>>>(enc, a2_wv, a2_bv, v2b, T_TOK, DIM, DIM);
  k_flash<<<dim3(S_TOK / 16, HEADS), 32, 0, stream>>>(qb, k2b, v2b, a4b, T_TOK, scale);
  // analysis
  k_qk_logits<<<dim3(S_TOK / 16, T_TOK / 16, HEADS), 32, 0, stream>>>(k2b, qb, logits, scale);
  k_argmax<<<(HEADS * T_TOK * FN + 255) / 256, 256, 0, stream>>>(logits, vidx);
  k_cross_softmax<<<S_TOK, 256, 0, stream>>>(logits, out_cm);
  k_energy<<<(HEADS * S_TOK + 255) / 256, 256, 0, stream>>>(a4b, Eb);
  k_headscore<<<1, 256, 0, stream>>>(Eb, flags);
  k_imap<<<dim3((PP + 15) / 16, T_TOK / 16, HEADS * FN), 32, 0, stream>>>(a4b, vidx, imap);
  k_imap_norm<<<(HEADS * S_TOK + 255) / 256, 256, 0, stream>>>(imap);
  k_imap_reduce<<<(T_TOK * S_TOK + 255) / 256, 256, 0, stream>>>(imap, flags, out_sel, out_sum);
  // cross-attn output projection + residual
  k_gemm<0><<<gemm_grid(S_TOK, DIM), 128, 0, stream>>>(a4b, a2_wo, a2_bo, tmpb, S_TOK, DIM, DIM);
  k_add_gate<<<(nSD4 + 255) / 256, 256, 0, stream>>>((const float4*)hsb, (const float4*)tmpb,
                                                     nullptr, (float4*)hsb, nSD4);
  // FFN
  k_ln<<<S_TOK, 256, 0, stream>>>(hsb, xb, nullptr, nullptr, mods + 4 * DIM, mods + 3 * DIM);
  k_gemm<1><<<gemm_grid(S_TOK, FFN_DIM), 128, 0, stream>>>(xb, ffw1, ffb1, ffnmid, S_TOK, FFN_DIM, DIM);
  k_gemm<0><<<gemm_grid(S_TOK, DIM), 128, 0, stream>>>(ffnmid, ffw2, ffb2, tmpb, S_TOK, DIM, FFN_DIM);
  k_add_gate<<<(nSD4 + 255) / 256, 256, 0, stream>>>((const float4*)hsb, (const float4*)tmpb,
                                                     mods + 5 * DIM, (float4*)out_hs, nSD4);
}
